// glimpse_network_58222576664632
// MI455X (gfx1250) — matrix operand order repaired and checked by reading, not compiled, not run on hardware
//
#include <hip/hip_runtime.h>
#include <hip/hip_bf16.h>

typedef __attribute__((ext_vector_type(16))) _Float16 v16h;
typedef __attribute__((ext_vector_type(8)))  _Float16 v8h;
typedef __attribute__((ext_vector_type(8)))  float    v8f;
typedef __attribute__((ext_vector_type(2)))  float    v2f;

// ---------------------------------------------------------------------------
// Kernel 1: foveate. One block per batch sample. Extract K=3 patches
// (16,32,64) centered at denormalized l, zero-padded, avg-pooled to 16x16,
// written channel-concatenated as f16: phi_h[b][i*3+c][gy*16+gx].
// ---------------------------------------------------------------------------
__global__ void __launch_bounds__(256)
foveate_kernel(const float* __restrict__ x, const float* __restrict__ l,
               _Float16* __restrict__ phi_h) {
    const int b   = blockIdx.x;
    const int tid = threadIdx.x;
    const float l0 = l[b * 2 + 0];
    const float l1 = l[b * 2 + 1];
    int c0 = (int)floorf((l0 + 1.0f) * 0.5f * 512.0f);
    int c1 = (int)floorf((l1 + 1.0f) * 0.5f * 512.0f);
    // dynamic_slice clamp: start in [0, padded - P] = [0, 512]
    c0 = min(max(c0, 0), 512);
    c1 = min(max(c1, 0), 512);
    const float* xb = x + (size_t)b * 3 * 512 * 512;
    for (int i = 0; i < 3; ++i) {
        const int P = 16 << i;        // patch size
        const int r = 1 << i;         // pool factor
        const int off = P >> 1;       // pad amount
        const float inv = 1.0f / (float)(r * r);
        for (int t = tid; t < 3 * 256; t += 256) {
            const int c  = t >> 8;
            const int g  = t & 255;
            const int gy = g >> 4, gx = g & 15;
            float s = 0.0f;
            for (int dy = 0; dy < r; ++dy) {
                const int y = c0 + gy * r + dy - off;
                if (y < 0 || y >= 512) continue;
                for (int dx = 0; dx < r; ++dx) {
                    const int xx = c1 + gx * r + dx - off;
                    if (xx < 0 || xx >= 512) continue;
                    s += xb[((size_t)c * 512 + y) * 512 + xx];
                }
            }
            phi_h[((size_t)b * 9 + i * 3 + c) * 256 + g] = (_Float16)(s * inv);
        }
    }
}

// ---------------------------------------------------------------------------
// Kernel 2: weight convert fp32 (Cout, Kin) -> f16 (Cout, Kp), zero padded.
// ---------------------------------------------------------------------------
__global__ void __launch_bounds__(256)
wconv_kernel(const float* __restrict__ w, _Float16* __restrict__ out,
             int Kin, int Kp, int total) {
    int idx = blockIdx.x * 256 + threadIdx.x;
    if (idx >= total) return;
    int co = idx / Kp;
    int k  = idx - co * Kp;
    out[idx] = (k < Kin) ? (_Float16)w[(size_t)co * Kin + k] : (_Float16)0.0f;
}

// ---------------------------------------------------------------------------
// Kernel 3: conv as implicit GEMM with v_wmma_f32_16x16x32_f16.
// One block per image (256 thr = 8 waves). Raw input (CIN,16,16) f16 staged
// in LDS once; then for each K-chunk of 32 the block cooperatively builds the
// 256x32 im2col slab in LDS, and each wave feeds its tiles' A-fragments with
// two ds_load_b128 each. Weights f16 [COUT][KP] read as contiguous 32B.
// ---------------------------------------------------------------------------
template <int CIN, int COUT, int KS, int PAD>
__global__ void __launch_bounds__(256)
conv_wmma_kernel(const _Float16* __restrict__ act_in,
                 const _Float16* __restrict__ wB,
                 const float*    __restrict__ bias,
                 _Float16* __restrict__ out_h, float* __restrict__ out_f) {
    constexpr int KK2   = KS * KS;
    constexpr int KREAL = CIN * KK2;
    constexpr int KP    = (KREAL + 31) & ~31;
    constexpr int NT    = COUT / 16;       // N tiles
    constexpr int TILES = NT * 16;         // x 16 M tiles
    constexpr int TPW   = TILES / 8;       // tiles per wave (8, 8, 16)

    __shared__ _Float16 sact[CIN * 256];   // raw feature maps
    __shared__ _Float16 scol[256 * 32];    // im2col chunk: [pixel m][klocal]

    const int b    = blockIdx.x;
    const int tid  = threadIdx.x;
    const int wave = tid >> 5;
    const int lane = tid & 31;
    const int lo   = lane & 15;
    const int hi   = lane >> 4;

    const _Float16* src = act_in + (size_t)b * (CIN * 256);
    for (int idx = tid; idx < CIN * 256; idx += 256) sact[idx] = src[idx];

    v8f acc[TPW] = {};
    const int ypix = tid >> 4, xpix = tid & 15;   // this thread's im2col row

    for (int kk = 0; kk < KP; kk += 32) {
        __syncthreads();                  // sact ready / previous chunk consumed
        // --- build im2col slab: thread tid owns pixel m=tid, all 32 klocal ---
#pragma unroll
        for (int e = 0; e < 32; ++e) {
            const int k = kk + e;
            _Float16 v = (_Float16)0.0f;
            if (k < KREAL) {
                const int ci  = k / KK2;          // compile-time KK2 -> mul/shift
                const int rem = k - ci * KK2;
                const int ky  = rem / KS;
                const int kx  = rem - ky * KS;
                const int yy  = ypix + ky - PAD;
                const int xc  = xpix + kx - PAD;
                if ((unsigned)yy < 16u && (unsigned)xc < 16u)
                    v = sact[ci * 256 + yy * 16 + xc];
            }
            scol[tid * 32 + e] = v;
        }
        __syncthreads();
        // --- consume: each wave's tiles do one wmma k-step ---
#pragma unroll
        for (int j = 0; j < TPW; ++j) {
            const int t     = wave + 8 * j;
            const int mtile = t & 15;
            const int ntile = t >> 4;
            const int m     = mtile * 16 + lo;
            // A layout (contiguous, matches B): k = hi*16 + e
            const v8h r0 = *(const v8h*)(scol + m * 32 + hi * 16);
            const v8h r1 = *(const v8h*)(scol + m * 32 + hi * 16 + 8);
            v16h a;
#pragma unroll
            for (int e = 0; e < 8; ++e) { a[e] = r0[e]; a[8 + e] = r1[e]; }
            // B fragment: column co = ntile*16+lo, k = hi*16+e (contiguous)
            const v16h bf = *(const v16h*)(wB + (size_t)(ntile * 16 + lo) * KP +
                                           kk + hi * 16);
            acc[j] = __builtin_amdgcn_wmma_f32_16x16x32_f16(
                false, a, false, bf, (short)0, acc[j], false, false);
        }
    }

    // D layout: element r -> m_local = r + 8*hi, n_local = lo
#pragma unroll
    for (int j = 0; j < TPW; ++j) {
        const int t     = wave + 8 * j;
        const int mtile = t & 15;
        const int ntile = t >> 4;
        const int co    = ntile * 16 + lo;
        const float bia = bias[co];
#pragma unroll
        for (int r2 = 0; r2 < 8; ++r2) {
            const int mo = mtile * 16 + r2 + 8 * hi;
            float v = acc[j][r2] + bia;
            v = v > 0.0f ? v : 0.0f;
            const size_t o = ((size_t)b * COUT + co) * 256 + mo;
            if (out_f) out_f[o] = v;
            else       out_h[o] = (_Float16)v;
        }
    }
}

// ---------------------------------------------------------------------------
// Kernel 4: where1 = relu(l @ loc1_w.T + b). K=2, pure VALU.
// ---------------------------------------------------------------------------
__global__ void __launch_bounds__(256)
where1_kernel(const float* __restrict__ l, const float* __restrict__ w,
              const float* __restrict__ bias, float* __restrict__ out) {
    const int idx = blockIdx.x * 256 + threadIdx.x;   // 64*1024
    const int b = idx >> 10;
    const int n = idx & 1023;
    float v = l[b * 2] * w[n * 2] + l[b * 2 + 1] * w[n * 2 + 1] + bias[n];
    out[idx] = v > 0.0f ? v : 0.0f;
}

// ---------------------------------------------------------------------------
// Kernel 5: fp32 GEMM with v_wmma_f32_16x16x4_f32.
// out(64,N) = relu(A(64,K) @ Bw(N,K)^T + bias) [* mul]. One block per N-tile
// of 16; 8 waves split K, partials reduced through LDS. Streams Bw once in
// fp32 (bandwidth-optimal for the 134MB fc_w).
// ---------------------------------------------------------------------------
__global__ void __launch_bounds__(256)
gemm_f32_wmma_kernel(const float* __restrict__ A, const float* __restrict__ Bw,
                     const float* __restrict__ bias, const float* __restrict__ mul,
                     float* __restrict__ out, int K, int N) {
    __shared__ float red[8 * 1024];
    const int nblk = blockIdx.x;
    const int tid  = threadIdx.x;
    const int wave = tid >> 5;
    const int lane = tid & 31;
    const int lo   = lane & 15;
    const int hi   = lane >> 4;
    const int kchunk = K >> 3;          // K divisible by 32 (1024, 32768)
    const int k0 = wave * kchunk;

    v8f acc0 = {}, acc1 = {}, acc2 = {}, acc3 = {};
    const float* brow = Bw + (size_t)(nblk * 16 + lo) * K;
    const float* a0   = A + (size_t)(0 * 16 + lo) * K;
    const float* a1   = A + (size_t)(1 * 16 + lo) * K;
    const float* a2   = A + (size_t)(2 * 16 + lo) * K;
    const float* a3   = A + (size_t)(3 * 16 + lo) * K;

    for (int k = k0; k < k0 + kchunk; k += 4) {
        const int ko = k + hi * 2;
        __builtin_prefetch(brow + k + 64, 0, 1);   // global_prefetch_b8
        v2f bf; bf[0] = brow[ko]; bf[1] = brow[ko + 1];
        v2f af;
        af[0] = a0[ko]; af[1] = a0[ko + 1];
        acc0 = __builtin_amdgcn_wmma_f32_16x16x4_f32(false, af, false, bf, (short)0, acc0, false, false);
        af[0] = a1[ko]; af[1] = a1[ko + 1];
        acc1 = __builtin_amdgcn_wmma_f32_16x16x4_f32(false, af, false, bf, (short)0, acc1, false, false);
        af[0] = a2[ko]; af[1] = a2[ko + 1];
        acc2 = __builtin_amdgcn_wmma_f32_16x16x4_f32(false, af, false, bf, (short)0, acc2, false, false);
        af[0] = a3[ko]; af[1] = a3[ko + 1];
        acc3 = __builtin_amdgcn_wmma_f32_16x16x4_f32(false, af, false, bf, (short)0, acc3, false, false);
    }
    // stash partials: idx = (mtile*16 + m_local)*16 + n_local
#pragma unroll
    for (int r = 0; r < 8; ++r) {
        const int ml = r + 8 * hi;
        red[wave * 1024 + (0 * 16 + ml) * 16 + lo] = acc0[r];
        red[wave * 1024 + (1 * 16 + ml) * 16 + lo] = acc1[r];
        red[wave * 1024 + (2 * 16 + ml) * 16 + lo] = acc2[r];
        red[wave * 1024 + (3 * 16 + ml) * 16 + lo] = acc3[r];
    }
    __syncthreads();
#pragma unroll
    for (int i = 0; i < 4; ++i) {
        const int idx = tid * 4 + i;        // 0..1023 = m*16 + nl
        float s = 0.0f;
#pragma unroll
        for (int w = 0; w < 8; ++w) s += red[w * 1024 + idx];
        const int m  = idx >> 4;
        const int nl = idx & 15;
        const int n  = nblk * 16 + nl;
        float v = s + bias[n];
        v = v > 0.0f ? v : 0.0f;
        if (mul) v *= mul[(size_t)m * N + n];
        out[(size_t)m * N + n] = v;
    }
}

// ---------------------------------------------------------------------------
extern "C" void kernel_launch(void* const* d_in, const int* in_sizes, int n_in,
                              void* d_out, int out_size, void* d_ws, size_t ws_size,
                              hipStream_t stream) {
    (void)in_sizes; (void)n_in; (void)out_size; (void)ws_size;
    const float* x       = (const float*)d_in[0];
    const float* l_prev  = (const float*)d_in[1];
    const float* conv1_w = (const float*)d_in[2];
    const float* conv1_b = (const float*)d_in[3];
    const float* conv2_w = (const float*)d_in[4];
    const float* conv2_b = (const float*)d_in[5];
    const float* conv3_w = (const float*)d_in[6];
    const float* conv3_b = (const float*)d_in[7];
    const float* fc_w    = (const float*)d_in[8];
    const float* fc_b    = (const float*)d_in[9];
    const float* loc1_w  = (const float*)d_in[10];
    const float* loc1_b  = (const float*)d_in[11];
    const float* loc2_w  = (const float*)d_in[12];
    const float* loc2_b  = (const float*)d_in[13];
    float* out = (float*)d_out;

    char* ws = (char*)d_ws;
    _Float16* phi_h = (_Float16*)(ws + 0);                 // 64*9*256*2   = 294912
    _Float16* w1h   = (_Float16*)(ws + 294912);            // 64*256*2     = 32768
    _Float16* w2h   = (_Float16*)(ws + 327680);            // 64*576*2     = 73728
    _Float16* w3h   = (_Float16*)(ws + 401408);            // 128*576*2    = 147456
    _Float16* act1  = (_Float16*)(ws + 548864);            // 64*64*256*2  = 2097152
    _Float16* act2  = (_Float16*)(ws + 2646016);           // 64*64*256*2  = 2097152
    float*    act3f = (float*)   (ws + 4743168);           // 64*128*256*4 = 8388608
    float*    wh1   = (float*)   (ws + 13131776);          // 64*1024*4    = 262144
    float*    wh2   = (float*)   (ws + 13393920);          // 64*1024*4    = 262144

    // 1) foveate -> phi_h (f16)
    foveate_kernel<<<64, 256, 0, stream>>>(x, l_prev, phi_h);

    // 2) weight conversions (Kp padded to multiple of 32)
    wconv_kernel<<<(64 * 256 + 255) / 256, 256, 0, stream>>>(conv1_w, w1h, 225, 256, 64 * 256);
    wconv_kernel<<<(64 * 576 + 255) / 256, 256, 0, stream>>>(conv2_w, w2h, 576, 576, 64 * 576);
    wconv_kernel<<<(128 * 576 + 255) / 256, 256, 0, stream>>>(conv3_w, w3h, 576, 576, 128 * 576);

    // 3) convs (implicit GEMM, v_wmma_f32_16x16x32_f16, chunked im2col in LDS)
    conv_wmma_kernel<9, 64, 5, 2><<<64, 256, 0, stream>>>(phi_h, w1h, conv1_b, act1, nullptr);
    conv_wmma_kernel<64, 64, 3, 1><<<64, 256, 0, stream>>>(act1, w2h, conv2_b, act2, nullptr);
    conv_wmma_kernel<64, 128, 3, 1><<<64, 256, 0, stream>>>(act2, w3h, conv3_b, nullptr, act3f);

    // 4) location path
    where1_kernel<<<256, 256, 0, stream>>>(l_prev, loc1_w, loc1_b, wh1);
    gemm_f32_wmma_kernel<<<64, 256, 0, stream>>>(wh1, loc2_w, loc2_b, nullptr, wh2, 1024, 1024);

    // 5) FC (streams fc_w once, fp32) fused with * where -> d_out
    gemm_f32_wmma_kernel<<<64, 256, 0, stream>>>(act3f, fc_w, fc_b, wh2, out, 32768, 1024);
}
